// SynthesisLayer_91268055040375
// MI455X (gfx1250) — matrix operand order repaired and checked by reading, not compiled, not run on hardware
//
#include <hip/hip_runtime.h>
#include <math.h>

typedef __attribute__((ext_vector_type(16))) _Float16 v16h;
typedef __attribute__((ext_vector_type(8)))  float    v8f;

#define N_  16
#define H_  64
#define W_  64
#define C_  512
#define O_  512
#define WD_ 512
#define NW_ 14

#define WSCALE   0.014731391274719739f   /* 1/sqrt(3*3*512) */
#define SSCALE   0.04419417382415922f    /* 1/sqrt(512)     */
#define SQRT2    1.4142135623730951f

// ---------------- kernel 1: style modulation s[n][c] ----------------
__global__ void style_kernel(const float* __restrict__ dlat,
                             const float* __restrict__ sw,
                             const float* __restrict__ sb,
                             const int*  __restrict__ li,
                             float* __restrict__ s) {
  const int n = blockIdx.x;
  const int c = threadIdx.x;
  const float* dl = dlat + ((long)n * NW_ + *li) * WD_;
  float acc = 0.f;
  for (int w = 0; w < WD_; ++w)
    acc += dl[w] * sw[(long)w * C_ + c];          // dl[w] scalar-broadcast
  s[n * C_ + c] = acc * SSCALE + sb[c];
}

// ---------------- kernel 2: xh[n,h,w,c] = f16(x * s) ----------------
__global__ void modx_kernel(const float* __restrict__ x,
                            const float* __restrict__ s,
                            _Float16* __restrict__ xh) {
  const long base = ((long)blockIdx.x * 256 + threadIdx.x) * 8;
  const int  c = (int)(base & (C_ - 1));
  const int  n = (int)(base >> 21);                // H*W*C = 2^21
  const float4* p = reinterpret_cast<const float4*>(x + base);
  float4 v0 = p[0], v1 = p[1];
  const float* sv = s + n * C_ + c;
  _Float16 hbuf[8];
  hbuf[0] = (_Float16)(v0.x * sv[0]); hbuf[1] = (_Float16)(v0.y * sv[1]);
  hbuf[2] = (_Float16)(v0.z * sv[2]); hbuf[3] = (_Float16)(v0.w * sv[3]);
  hbuf[4] = (_Float16)(v1.x * sv[4]); hbuf[5] = (_Float16)(v1.y * sv[5]);
  hbuf[6] = (_Float16)(v1.z * sv[6]); hbuf[7] = (_Float16)(v1.w * sv[7]);
  *reinterpret_cast<float4*>(xh + base) = *reinterpret_cast<float4*>(hbuf);
}

// ------- kernel 3: wh[(ky,kx), o, c] = f16(w * WSCALE), transposed ---
__global__ void modw_kernel(const float* __restrict__ wt,
                            _Float16* __restrict__ wh) {
  const long base = ((long)blockIdx.x * 256 + threadIdx.x) * 8;  // over [kk][c][o]
  const int  o  = (int)(base & (O_ - 1));
  const int  c  = (int)((base >> 9) & (C_ - 1));
  const int  kk = (int)(base >> 18);
  const float4* p = reinterpret_cast<const float4*>(wt + base);
  float4 v0 = p[0], v1 = p[1];
  const long obase = ((long)kk * O_ + o) * C_ + c;
  wh[obase + 0 * C_] = (_Float16)(v0.x * WSCALE);
  wh[obase + 1 * C_] = (_Float16)(v0.y * WSCALE);
  wh[obase + 2 * C_] = (_Float16)(v0.z * WSCALE);
  wh[obase + 3 * C_] = (_Float16)(v0.w * WSCALE);
  wh[obase + 4 * C_] = (_Float16)(v1.x * WSCALE);
  wh[obase + 5 * C_] = (_Float16)(v1.y * WSCALE);
  wh[obase + 6 * C_] = (_Float16)(v1.z * WSCALE);
  wh[obase + 7 * C_] = (_Float16)(v1.w * WSCALE);
}

// ---------------- kernel 4: wsum[i][o] = scale^2 * sum_hw w^2 -------
__global__ void wsum_kernel(const float* __restrict__ wt,
                            float* __restrict__ wsum) {
  const int idx = blockIdx.x * blockDim.x + threadIdx.x;   // 512*512
  const int o = idx & (O_ - 1);
  const int i = idx >> 9;
  float acc = 0.f;
  for (int t = 0; t < 9; ++t) {
    float v = wt[((long)t * C_ + i) * O_ + o];
    acc += v * v;
  }
  wsum[idx] = acc * (WSCALE * WSCALE);
}

// ---------------- kernel 5: demod d[n][o] ---------------------------
__global__ void demod_kernel(const float* __restrict__ wsum,
                             const float* __restrict__ s,
                             float* __restrict__ d) {
  const int idx = blockIdx.x * blockDim.x + threadIdx.x;   // 16*512
  const int o = idx & (O_ - 1);
  const int n = idx >> 9;
  float acc = 1e-8f;
  for (int i = 0; i < C_; ++i) {
    float sv = s[n * C_ + i];
    acc += wsum[(long)i * O_ + o] * sv * sv;
  }
  d[idx] = rsqrtf(acc);
}

// ---------------- kernel 6: conv as implicit WMMA GEMM --------------
// Block: 256 threads (8 waves). Tile: 64 pixels (one row) x 128 out-ch.
// A staged with halo (66 px) once per (ky, c-step), reused across kx.
// Double-buffered LDS, one barrier per step, 12 WMMA / wave / step.
__global__ __launch_bounds__(256) void conv_kernel(
    const _Float16* __restrict__ xh,  const _Float16* __restrict__ wh,
    const float* __restrict__ dmod,   const float* __restrict__ bias,
    const float* __restrict__ noise,  const float* __restrict__ nstr,
    float* __restrict__ out) {

  __shared__ __align__(16) _Float16 Alds[2][66][40];       // [halo-px][k]
  __shared__ __align__(16) _Float16 Blds[2][3][128][40];   // [kx][o][k]

  const int n  = blockIdx.z;
  const int h  = blockIdx.y;
  const int o0 = blockIdx.x * 128;
  const int t  = threadIdx.x;
  const int lane = t & 31, wv = t >> 5;
  const int mi = wv & 3;           // 16-pixel M tile
  const int nq = wv >> 2;          // covers o-range nq*64 .. nq*64+63

  v8f acc[4] = {{}, {}, {}, {}};

  auto stage = [&](int step, int buf) {
    const int ky = step >> 4;
    const int c0 = (step & 15) * 32;
    const int hy = h + ky - 1;
    // A: 66 halo rows x 32 halves  = 264 chunks of 16B
#pragma unroll
    for (int i = 0; i < 2; ++i) {
      const int idx = t + 256 * i;
      if (idx < 264) {
        const int r = idx >> 2, ch = (idx & 3) * 8;
        const int wx = r - 1;
        float4 v = {0.f, 0.f, 0.f, 0.f};
        if (hy >= 0 && hy < H_ && wx >= 0 && wx < W_)
          v = *reinterpret_cast<const float4*>(
              xh + (((long)n * H_ + hy) * W_ + wx) * C_ + c0 + ch);
        *reinterpret_cast<float4*>(&Alds[buf][r][ch]) = v;
      }
    }
    // B: 3 kx x 128 o x 32 c = 1536 chunks of 16B
#pragma unroll
    for (int i = 0; i < 6; ++i) {
      const int idx = t + 256 * i;
      const int r = idx >> 2, ch = (idx & 3) * 8;
      const int kx = r >> 7, ol = r & 127;
      float4 v = *reinterpret_cast<const float4*>(
          wh + ((long)((ky * 3 + kx) * O_ + o0 + ol)) * C_ + c0 + ch);
      *reinterpret_cast<float4*>(&Blds[buf][kx][ol][ch]) = v;
    }
  };

  const int mrow  = mi * 16 + (lane & 15);
  const int khalf = lane >> 4;

  stage(0, 0);
  __syncthreads();

  for (int step = 0; step < 48; ++step) {
    const int buf = step & 1;
    if (step + 1 < 48) stage(step + 1, buf ^ 1);
#pragma unroll
    for (int kx = 0; kx < 3; ++kx) {
      v16h af;
#pragma unroll
      for (int j = 0; j < 8; ++j) {            // ISA 16-bit A layout
        const int k = (j >> 2) * 16 + khalf * 8 + (j & 3) * 2;
        af[2 * j]     = Alds[buf][mrow + kx][k];
        af[2 * j + 1] = Alds[buf][mrow + kx][k + 1];
      }
#pragma unroll
      for (int tt = 0; tt < 4; ++tt) {
        const int ol = nq * 64 + tt * 16 + (lane & 15);
        v16h bf;
#pragma unroll
        for (int j = 0; j < 8; ++j) {          // ISA 16-bit B layout (matches A)
          const int k = (j >> 2) * 16 + khalf * 8 + (j & 3) * 2;
          bf[2 * j]     = Blds[buf][kx][ol][k];
          bf[2 * j + 1] = Blds[buf][kx][ol][k + 1];
        }
        acc[tt] = __builtin_amdgcn_wmma_f32_16x16x32_f16(
                      false, af, false, bf, (short)0, acc[tt], false, false);
      }
    }
    __syncthreads();
  }

  // ---- epilogue: demod, noise, bias, leaky-relu * sqrt(2) ----
  const float ns = *nstr;
  const long pixbase = ((long)n * H_ + h) * W_;
#pragma unroll
  for (int tt = 0; tt < 4; ++tt) {
    const int oc = o0 + nq * 64 + tt * 16 + (lane & 15);
    const float dv = dmod[n * O_ + oc];
    const float bv = bias[oc];
#pragma unroll
    for (int j = 0; j < 8; ++j) {
      const int  wpix = mi * 16 + khalf * 8 + j;   // C/D: VGPR j -> M row
      const long pix  = pixbase + wpix;
      float v = acc[tt][j] * dv + noise[pix] * ns + bv;
      v = (v > 0.f ? v : 0.2f * v) * SQRT2;
      out[pix * O_ + oc] = v;
    }
  }
}

// --------------------------------------------------------------------
extern "C" void kernel_launch(void* const* d_in, const int* in_sizes, int n_in,
                              void* d_out, int out_size, void* d_ws, size_t ws_size,
                              hipStream_t stream) {
  const float* x      = (const float*)d_in[0];
  const float* dlat   = (const float*)d_in[1];
  const float* styw   = (const float*)d_in[2];
  const float* styb   = (const float*)d_in[3];
  const float* weight = (const float*)d_in[4];
  const float* bias   = (const float*)d_in[5];
  const float* nstr   = (const float*)d_in[6];
  const float* noise  = (const float*)d_in[7];
  const int*   lidx   = (const int*)d_in[8];
  float*       out    = (float*)d_out;

  _Float16* xh = (_Float16*)d_ws;                     // 16*64*64*512 halves
  _Float16* wh = xh + (size_t)N_ * H_ * W_ * C_;      // 9*512*512 halves
  float* s     = (float*)(wh + (size_t)9 * O_ * C_);  // 16*512
  float* wsum  = s + N_ * C_;                         // 512*512
  float* dm    = wsum + C_ * O_;                      // 16*512

  style_kernel<<<N_, C_, 0, stream>>>(dlat, styw, styb, lidx, s);
  modx_kernel<<<(N_ * H_ * W_ * C_) / (256 * 8), 256, 0, stream>>>(x, s, xh);
  modw_kernel<<<(9 * C_ * O_) / (256 * 8), 256, 0, stream>>>(weight, wh);
  wsum_kernel<<<(C_ * O_) / 256, 256, 0, stream>>>(weight, wsum);
  demod_kernel<<<(N_ * O_) / 256, 256, 0, stream>>>(wsum, s, dm);

  dim3 grid(O_ / 128, H_, N_);
  conv_kernel<<<grid, 256, 0, stream>>>(xh, wh, dm, bias, noise, nstr, out);
}
